// SparseLinearAttention_33328946217069
// MI455X (gfx1250) — matrix operand order repaired and checked by reading, not compiled, not run on hardware
//
#include <hip/hip_runtime.h>

// ---------------------------------------------------------------------------
// SparseLinearAttention for MI455X (gfx1250, wave32, WMMA bf16 16x16x32)
// B=2,H=16,L=4096,D=128, BLKQ=BLKK=64, nqb=nkb=64, topk=16
// ---------------------------------------------------------------------------

typedef __bf16 bf16_t;
typedef __attribute__((ext_vector_type(16))) __bf16 v16bf;
typedef __attribute__((ext_vector_type(8)))  __bf16 v8bf;
typedef __attribute__((ext_vector_type(8)))  float  v8f;

#define DD     128
#define BLK    64
#define NQB    64
#define NKB    64
#define TOPK   16
#define SEQ    4096
#define EPSF   1e-6f
#define VTS    72      // padded row-stride (elems) of transposed-v LDS tile:
                       // 72*2B = 144B, multiple of 16 -> b128-aligned columns

// ---------------------------------------------------------------------------
// WMMA helper
// ---------------------------------------------------------------------------
__device__ __forceinline__ v8f wmma_bf16(v16bf a, v16bf b, v8f c) {
  return __builtin_amdgcn_wmma_f32_16x16x32_bf16(
      /*neg_a=*/false, a, /*neg_b=*/false, b,
      /*c_mod=*/(short)0, c, /*reuse_a=*/false, /*reuse_b=*/false);
}

// A fragment: 16x32 bf16 tile (MxK), row-major LDS source, stride in elems.
// Lane l: m = l&15, hi = l>>4; VGPR v holds K pair:
//   k = v*2 + half + hi*16   (one contiguous 16-elem run)
__device__ __forceinline__ v16bf load_A(const bf16_t* src, int stride,
                                        int row0, int col0, int lane) {
  int m = lane & 15, hi = lane >> 4;
  const bf16_t* p = src + (row0 + m) * stride + col0 + hi * 16;
  v16bf a;
#pragma unroll
  for (int v = 0; v < 8; ++v) {
    int k = v * 2;
    a[2 * v]     = p[k];
    a[2 * v + 1] = p[k + 1];
  }
  return a;
}

// B fragment for logical B[k][n] = src[n][k] (src row-major, row = n).
// Lane l: n = l&15, hi = l>>4; element e -> k = kbase + hi*16 + e (contig).
__device__ __forceinline__ v16bf load_B_T(const bf16_t* src, int stride,
                                          int kbase, int ncol0, int lane) {
  int n = lane & 15, hi = lane >> 4;
  const bf16_t* p = src + (ncol0 + n) * stride + kbase + hi * 16;
  v16bf b;
#pragma unroll
  for (int e = 0; e < 16; ++e) b[e] = p[e];
  return b;
}

// ---------------------------------------------------------------------------
// Row softmax of a 64x128 fp32 block -> bf16 LDS (4 waves x 16 rows each).
// ---------------------------------------------------------------------------
__device__ __forceinline__ void softmax_block(const float* __restrict__ src,
                                              bf16_t* dst, int wave, int lane) {
#pragma unroll 1
  for (int i = 0; i < 16; ++i) {
    int row = wave * 16 + i;
    float4 x = ((const float4*)(src + row * DD))[lane];
    float m = fmaxf(fmaxf(x.x, x.y), fmaxf(x.z, x.w));
#pragma unroll
    for (int off = 16; off >= 1; off >>= 1) m = fmaxf(m, __shfl_xor(m, off, 32));
    float e0 = __expf(x.x - m), e1 = __expf(x.y - m);
    float e2 = __expf(x.z - m), e3 = __expf(x.w - m);
    float s = e0 + e1 + e2 + e3;
#pragma unroll
    for (int off = 16; off >= 1; off >>= 1) s += __shfl_xor(s, off, 32);
    float inv = 1.0f / s;
    bf16_t* d = dst + row * DD + 4 * lane;
    d[0] = (bf16_t)(e0 * inv);
    d[1] = (bf16_t)(e1 * inv);
    d[2] = (bf16_t)(e2 * inv);
    d[3] = (bf16_t)(e3 * inv);
  }
}

// Same, plus per-lane accumulation of column sums (normalizer Z):
// lane owns columns 4*lane..4*lane+3 of its wave's 16 rows.
__device__ __forceinline__ void softmax_block_z(const float* __restrict__ src,
                                                bf16_t* dst, float z4[4],
                                                int wave, int lane) {
#pragma unroll 1
  for (int i = 0; i < 16; ++i) {
    int row = wave * 16 + i;
    float4 x = ((const float4*)(src + row * DD))[lane];
    float m = fmaxf(fmaxf(x.x, x.y), fmaxf(x.z, x.w));
#pragma unroll
    for (int off = 16; off >= 1; off >>= 1) m = fmaxf(m, __shfl_xor(m, off, 32));
    float e0 = __expf(x.x - m), e1 = __expf(x.y - m);
    float e2 = __expf(x.z - m), e3 = __expf(x.w - m);
    float s = e0 + e1 + e2 + e3;
#pragma unroll
    for (int off = 16; off >= 1; off >>= 1) s += __shfl_xor(s, off, 32);
    float inv = 1.0f / s;
    bf16_t b0 = (bf16_t)(e0 * inv), b1 = (bf16_t)(e1 * inv);
    bf16_t b2 = (bf16_t)(e2 * inv), b3 = (bf16_t)(e3 * inv);
    bf16_t* d = dst + row * DD + 4 * lane;
    d[0] = b0; d[1] = b1; d[2] = b2; d[3] = b3;
    z4[0] += (float)b0; z4[1] += (float)b1;
    z4[2] += (float)b2; z4[3] += (float)b3;
  }
}

// fp32 64x128 block -> bf16 *transposed* LDS tile vt[col*VTS + row].
// Each lane buffers 8 rows of its 4 columns and writes one b128 per column.
__device__ __forceinline__ void cvt_block_t(const float* __restrict__ src,
                                            bf16_t* dstT, int wave, int lane) {
#pragma unroll 1
  for (int half = 0; half < 2; ++half) {
    int r0 = wave * 16 + half * 8;
    float xa[8][4];
#pragma unroll
    for (int j = 0; j < 8; ++j) {
      float4 t = ((const float4*)(src + (r0 + j) * DD))[lane];
      xa[j][0] = t.x; xa[j][1] = t.y; xa[j][2] = t.z; xa[j][3] = t.w;
    }
#pragma unroll
    for (int c = 0; c < 4; ++c) {
      v8bf p;
#pragma unroll
      for (int j = 0; j < 8; ++j) p[j] = (bf16_t)xa[j][c];
      *(v8bf*)(dstT + (4 * lane + c) * VTS + r0) = p;
    }
  }
}

// ---------------------------------------------------------------------------
// Kernel 1: per-block mean pooling for q and k.
// ---------------------------------------------------------------------------
__global__ __launch_bounds__(128) void block_means(
    const float* __restrict__ q, const float* __restrict__ k,
    float* __restrict__ qbm, float* __restrict__ kbm, int nblk) {
  int bid = blockIdx.x;
  const float* src;
  float* dst;
  if (bid < nblk) { src = q + (size_t)bid * BLK * DD; dst = qbm + (size_t)bid * DD; }
  else { int b2 = bid - nblk; src = k + (size_t)b2 * BLK * DD; dst = kbm + (size_t)b2 * DD; }
  int t = threadIdx.x;
  float s = 0.0f;
#pragma unroll 4
  for (int r = 0; r < BLK; ++r) s += src[r * DD + t];
  dst[t] = s * (1.0f / (float)BLK);
}

// ---------------------------------------------------------------------------
// Kernel 2: block scores + top-16 (rank-by-comparison, ties -> lower index,
// matching jax.lax.top_k ordering).
// ---------------------------------------------------------------------------
__global__ __launch_bounds__(64) void topk_sel(
    const float* __restrict__ qbm, const float* __restrict__ kbm,
    int* __restrict__ idx) {
  int bid = blockIdx.x;           // bh * NQB + qb
  int bh  = bid / NQB;
  __shared__ float qrow[DD];
  __shared__ float sc[NKB];
  int t = threadIdx.x;
  qrow[t]      = qbm[(size_t)bid * DD + t];
  qrow[t + 64] = qbm[(size_t)bid * DD + 64 + t];
  __syncthreads();
  const float* kp = kbm + ((size_t)bh * NKB + t) * DD;
  float s = 0.0f;
#pragma unroll 4
  for (int d = 0; d < DD; ++d) s += qrow[d] * kp[d];
  sc[t] = s;
  __syncthreads();
  float mys = sc[t];
  int rank = 0;
#pragma unroll 4
  for (int m = 0; m < NKB; ++m) {
    float sm = sc[m];
    rank += (sm > mys || (sm == mys && m < t)) ? 1 : 0;
  }
  if (rank < TOPK) idx[(size_t)bid * TOPK + rank] = t;
}

// ---------------------------------------------------------------------------
// Kernel 3 (core): per q-block sparse linear attention, bf16 WMMA.
// 128 threads = 4 waves; wave w owns rows 16w..16w+15.
// Per selected kv block:  S = cq @ ckT (64x64, K=128) ; O += S @ v (64x128, K=64)
// ---------------------------------------------------------------------------
__global__ __launch_bounds__(128) void sla_core(
    const float* __restrict__ q, const float* __restrict__ k,
    const float* __restrict__ v, const int* __restrict__ idx,
    float* __restrict__ out) {
  int bid = blockIdx.x;           // bh * NQB + qb
  int bh  = bid / NQB;

  __shared__ __align__(16) bf16_t lds_cq[BLK * DD];   // 16 KB
  __shared__ __align__(16) bf16_t lds_ck[BLK * DD];   // 16 KB
  __shared__ __align__(16) bf16_t lds_vt[DD * VTS];   // 18 KB (v transposed)
  __shared__ __align__(16) bf16_t lds_s [BLK * BLK];  //  8 KB (per-wave strips)
  __shared__ float  lds_zp[4 * DD];                   //  2 KB (per-wave Z parts)
  __shared__ float  lds_z [DD];
  __shared__ float  lds_den[BLK];

  int tid  = threadIdx.x;
  int wave = tid >> 5;
  int lane = tid & 31;

  const float* qp = q + (size_t)bid * BLK * DD;
  const int*  sel = idx + (size_t)bid * TOPK;

  softmax_block(qp, lds_cq, wave, lane);
  __syncthreads();

  v8f O[8] = {};          // 16x128 strip accumulator (fp32)
  float z4[4] = {};       // per-lane partial Z (cols 4*lane..4*lane+3)

  for (int t = 0; t < TOPK; ++t) {
    int n = sel[t];
    const float* kp = k + ((size_t)bh * SEQ + (size_t)n * BLK) * DD;
    const float* vp = v + ((size_t)bh * SEQ + (size_t)n * BLK) * DD;

    __syncthreads();      // previous-iteration readers of lds_ck/lds_vt done
    softmax_block_z(kp, lds_ck, z4, wave, lane);
    cvt_block_t(vp, lds_vt, wave, lane);
    __syncthreads();

    // Prefetch next selected kv block into cache while WMMAs run
    // (gfx1250 global_prefetch_b8; q/k/v are L2-resident at 192 MB total).
    if (t + 1 < TOPK) {
      int n2 = sel[t + 1];
      const char* kp2 = (const char*)(k + ((size_t)bh * SEQ + (size_t)n2 * BLK) * DD);
      const char* vp2 = (const char*)(v + ((size_t)bh * SEQ + (size_t)n2 * BLK) * DD);
      __builtin_prefetch(kp2 + tid * 128, 0, 3);
      __builtin_prefetch(kp2 + (tid + 128) * 128, 0, 3);
      __builtin_prefetch(vp2 + tid * 128, 0, 3);
      __builtin_prefetch(vp2 + (tid + 128) * 128, 0, 3);
    }

    // S strip = cq[16x128] @ ckT[128x64]
    v8f S[4] = {};
#pragma unroll
    for (int kk = 0; kk < DD; kk += 32) {
      v16bf a = load_A(lds_cq, DD, wave * 16, kk, lane);
#pragma unroll
      for (int j = 0; j < 4; ++j) {
        v16bf b = load_B_T(lds_ck, DD, kk, j * 16, lane);
        S[j] = wmma_bf16(a, b, S[j]);
      }
    }

    // D-layout -> row-major bf16 in per-wave LDS scratch (same-wave LDS ops
    // are in order; strips are disjoint so no barrier needed).
#pragma unroll
    for (int j = 0; j < 4; ++j) {
#pragma unroll
      for (int r = 0; r < 8; ++r) {
        int row = wave * 16 + r + 8 * (lane >> 4);
        int col = j * 16 + (lane & 15);
        lds_s[row * BLK + col] = (bf16_t)S[j][r];
      }
    }

    // O strip += S[16x64] @ v[64x128]  (v held transposed: contiguous b128)
#pragma unroll
    for (int kk = 0; kk < BLK; kk += 32) {
      v16bf a = load_A(lds_s, BLK, wave * 16, kk, lane);
#pragma unroll
      for (int j = 0; j < 8; ++j) {
        v16bf b = load_B_T(lds_vt, VTS, kk, j * 16, lane);
        O[j] = wmma_bf16(a, b, O[j]);
      }
    }
  }

  // Z: combine 4 per-wave partials, then den[l] = sum_d cq[l][d]*z[d] + eps
  __syncthreads();
#pragma unroll
  for (int c = 0; c < 4; ++c) lds_zp[wave * DD + 4 * lane + c] = z4[c];
  __syncthreads();
  {
    float zc = lds_zp[0 * DD + tid] + lds_zp[1 * DD + tid] +
               lds_zp[2 * DD + tid] + lds_zp[3 * DD + tid];
    lds_z[tid] = zc;
  }
  __syncthreads();
  if (tid < BLK) {
    float den = 0.0f;
#pragma unroll 4
    for (int d = 0; d < DD; ++d) den += (float)lds_cq[tid * DD + d] * lds_z[d];
    lds_den[tid] = den + EPSF;
  }
  __syncthreads();

  // out = O / den  (D fragment: row = r + 8*(lane>>4), col = lane&15)
  float* op = out + (size_t)bid * BLK * DD;
#pragma unroll
  for (int j = 0; j < 8; ++j) {
#pragma unroll
    for (int r = 0; r < 8; ++r) {
      int row = wave * 16 + r + 8 * (lane >> 4);
      int col = j * 16 + (lane & 15);
      op[row * DD + col] = O[j][r] / lds_den[row];
    }
  }
}

// ---------------------------------------------------------------------------
// Launch
// ---------------------------------------------------------------------------
extern "C" void kernel_launch(void* const* d_in, const int* in_sizes, int n_in,
                              void* d_out, int out_size, void* d_ws, size_t ws_size,
                              hipStream_t stream) {
  (void)n_in; (void)out_size; (void)ws_size;
  const float* q = (const float*)d_in[0];
  const float* k = (const float*)d_in[1];
  const float* v = (const float*)d_in[2];
  float* out = (float*)d_out;

  int BH   = in_sizes[0] / (SEQ * DD);   // = 32
  int nblk = BH * NQB;                   // = 2048

  float* qbm = (float*)d_ws;                       // nblk * 128 f32
  float* kbm = qbm + (size_t)nblk * DD;            // nblk * 128 f32
  int*   idx = (int*)(kbm + (size_t)nblk * DD);    // nblk * 16 i32

  block_means<<<2 * nblk, 128, 0, stream>>>(q, k, qbm, kbm, nblk);
  topk_sel   <<<nblk,      64, 0, stream>>>(qbm, kbm, idx);
  sla_core   <<<nblk,     128, 0, stream>>>(q, k, v, idx, out);
}
